// CENetCP_20444044329678
// MI455X (gfx1250) — compile-verified, matrix operand order checked by reading, not run on hardware
//
#include <hip/hip_runtime.h>
#include <hip/hip_bf16.h>
#include <math.h>

// ---------------- problem constants ----------------
#define NE   40000      // NUM_ENTS
#define NR   480        // NUM_RELS
#define HD   256        // H
#define NHD  4          // NH
#define DHD  64         // DH
#define LL   2          // L
#define GG   3          // G
#define EHE  30000      // EH (edges per hist graph)
#define QQ   2048       // Q
#define KTOP 50         // K
#define PP   2          // P
#define CC   50         // C
#define CH   (CC*HD)    // 12800
#define N2   (PP*NE)    // 80000
#define EC   (QQ*KTOP)  // 102400

typedef __attribute__((ext_vector_type(16))) _Float16 v16h;
typedef __attribute__((ext_vector_type(8)))  _Float16 v8h;
typedef __attribute__((ext_vector_type(4)))  _Float16 v4h;
typedef __attribute__((ext_vector_type(8)))  float    v8f;

enum { ACT_NONE = 0, ACT_RELU = 1, ACT_SIGMOID = 2 };
enum { ACC_NONE = 0, ACC_ADD = 1, ACC_BLEND = 2 };

static __host__ __device__ __forceinline__ int dmin(int a, int b) { return a < b ? a : b; }

// ---------------- generic WMMA GEMM:  C[M,N] = epilogue(A[M,K] @ B) ----------------
// BM=128 BN=64 BK=32, 256 threads = 8 wave32 waves.
// Each wave owns a 16-row stripe and all four 16-col tiles -> 4 WMMAs / k-step.
// Requirements (met by every call site): K % 32 == 0, N % 4 == 0,
// lda/ldb % 4 == 0, 16B-aligned base pointers. M,N ragged edges handled by
// clamped loads + guarded stores.
#define BM 128
#define BN 64
#define BK 32

__global__ __launch_bounds__(256)
void kGemmWMMA(const float* __restrict__ A, int lda,
               const float* __restrict__ B, int ldb, int transB,
               float* __restrict__ C, int ldc,
               const float* __restrict__ bias,
               int M, int N, int Kdim, int act, int accMode)
{
    __shared__ _Float16 As[BM * BK];   // row-major  [row][k]  128x32
    __shared__ _Float16 Bt[BN * BK];   // transposed [col][k]   64x32

    const int tid  = threadIdx.x;
    const int lane = tid & 31;
    const int wave = tid >> 5;         // 0..7 -> 16-row stripe
    const int m0   = blockIdx.y * BM;
    const int n0   = blockIdx.x * BN;

    v8f acc0 = {}, acc1 = {}, acc2 = {}, acc3 = {};

    const int arow = wave * 16 + (lane & 15);
    const int hf   = lane >> 4;        // which K-half this lane holds

    // staging coordinates (fixed per thread)
    int ars[4], acs[4];                // A slots
    #pragma unroll
    for (int j = 0; j < 4; ++j) {
        int s = tid + j * 256;
        ars[j] = s >> 3; acs[j] = s & 7;
    }

    for (int k0 = 0; k0 < Kdim; k0 += BK) {
        // ---- stage A tile: 4 b128 loads first (one clause), then cvt+b64 LDS stores
        float4 va[4];
        #pragma unroll
        for (int j = 0; j < 4; ++j) {
            int gm = dmin(m0 + ars[j], M - 1);
            va[j] = *reinterpret_cast<const float4*>(A + (size_t)gm * lda + k0 + acs[j] * 4);
        }
        // ---- stage B tile (loads batched likewise) ----
        float4 vb[2];
        if (transB) {
            #pragma unroll
            for (int j = 0; j < 2; ++j) {
                int s  = tid + j * 256;
                int c  = s >> 3, kg = s & 7;
                int gn = dmin(n0 + c, N - 1);
                vb[j] = *reinterpret_cast<const float4*>(B + (size_t)gn * ldb + k0 + kg * 4);
            }
        } else {
            #pragma unroll
            for (int j = 0; j < 2; ++j) {
                int s  = tid + j * 256;
                int k  = s >> 4, ng = s & 15;
                int gn = dmin(n0 + ng * 4, N - 4);
                vb[j] = *reinterpret_cast<const float4*>(B + (size_t)(k0 + k) * ldb + gn);
            }
        }

        // convert + store A
        #pragma unroll
        for (int j = 0; j < 4; ++j) {
            v4h h; h[0] = (_Float16)va[j].x; h[1] = (_Float16)va[j].y;
                   h[2] = (_Float16)va[j].z; h[3] = (_Float16)va[j].w;
            *reinterpret_cast<v4h*>(&As[ars[j] * BK + acs[j] * 4]) = h;
        }
        // convert + store B
        if (transB) {
            #pragma unroll
            for (int j = 0; j < 2; ++j) {
                int s = tid + j * 256;
                int c = s >> 3, kg = s & 7;
                v4h h; h[0] = (_Float16)vb[j].x; h[1] = (_Float16)vb[j].y;
                       h[2] = (_Float16)vb[j].z; h[3] = (_Float16)vb[j].w;
                *reinterpret_cast<v4h*>(&Bt[c * BK + kg * 4]) = h;
            }
        } else {
            #pragma unroll
            for (int j = 0; j < 2; ++j) {
                int s = tid + j * 256;
                int k = s >> 4, ng = s & 15;
                Bt[(ng * 4 + 0) * BK + k] = (_Float16)vb[j].x;
                Bt[(ng * 4 + 1) * BK + k] = (_Float16)vb[j].y;
                Bt[(ng * 4 + 2) * BK + k] = (_Float16)vb[j].z;
                Bt[(ng * 4 + 3) * BK + k] = (_Float16)vb[j].w;
            }
        }
        __syncthreads();

        // prefetch next-k tiles through the gfx1250 prefetch path
        if (k0 + BK < Kdim) {
            __builtin_prefetch(&A[(size_t)dmin(m0 + (tid >> 1), M - 1) * lda + (k0 + BK)], 0, 1);
            if (transB)
                __builtin_prefetch(&B[(size_t)dmin(n0 + (tid & 63), N - 1) * ldb + (k0 + BK)], 0, 1);
            else
                __builtin_prefetch(&B[(size_t)(k0 + BK) * ldb + dmin(n0 + (tid & 63), N - 1)], 0, 1);
        }

        // ---- A fragment: two contiguous 8-half chunks (2x ds_load_b128) ----
        // element i<8  -> kk = hf*8 + i ; element i>=8 -> kk = 16 + hf*8 + (i-8)
        v8h a_lo = *reinterpret_cast<const v8h*>(&As[arow * BK + hf * 8]);
        v8h a_hi = *reinterpret_cast<const v8h*>(&As[arow * BK + 16 + hf * 8]);
        v16h a = __builtin_shufflevector(a_lo, a_hi, 0,1,2,3,4,5,6,7,8,9,10,11,12,13,14,15);

        // ---- B fragments per 16-col tile, same chunked layout from Bt ----
        #pragma unroll
        for (int ct = 0; ct < 4; ++ct) {
            const int brow = ct * 16 + (lane & 15);
            v8h b_lo = *reinterpret_cast<const v8h*>(&Bt[brow * BK + hf * 8]);
            v8h b_hi = *reinterpret_cast<const v8h*>(&Bt[brow * BK + 16 + hf * 8]);
            v16h b = __builtin_shufflevector(b_lo, b_hi, 0,1,2,3,4,5,6,7,8,9,10,11,12,13,14,15);
            if      (ct == 0) acc0 = __builtin_amdgcn_wmma_f32_16x16x32_f16(false, a, false, b, (short)0, acc0, false, false);
            else if (ct == 1) acc1 = __builtin_amdgcn_wmma_f32_16x16x32_f16(false, a, false, b, (short)0, acc1, false, false);
            else if (ct == 2) acc2 = __builtin_amdgcn_wmma_f32_16x16x32_f16(false, a, false, b, (short)0, acc2, false, false);
            else              acc3 = __builtin_amdgcn_wmma_f32_16x16x32_f16(false, a, false, b, (short)0, acc3, false, false);
        }
        __syncthreads();
    }

    // ---- epilogue + store (C/D layout: lane%16 = col, VGPR v -> row v + 8*(lane/16)) ----
    #pragma unroll
    for (int ct = 0; ct < 4; ++ct) {
        const v8f acc = (ct == 0) ? acc0 : (ct == 1) ? acc1 : (ct == 2) ? acc2 : acc3;
        int gn = n0 + ct * 16 + (lane & 15);
        if (gn >= N) continue;
        float bv = bias ? bias[gn] : 0.0f;
        #pragma unroll
        for (int v = 0; v < 8; ++v) {
            int gm = m0 + wave * 16 + v + ((lane >> 4) << 3);
            if (gm >= M) continue;
            size_t o = (size_t)gm * ldc + gn;
            float val = acc[v] + bv;
            if (accMode == ACC_ADD) val += C[o];
            if (act == ACT_RELU)         val = fmaxf(val, 0.0f);
            else if (act == ACT_SIGMOID) val = 1.0f / (1.0f + __expf(-val));
            if (accMode == ACC_BLEND)    val = 0.5f * C[o] + 0.5f * val;
            C[o] = val;
        }
    }
}

// ---------------- elementwise / reduction helpers ----------------
__global__ __launch_bounds__(256)
void kRowNorm(const float* __restrict__ in, float* __restrict__ out, int rows)
{
    __shared__ float red[256];
    int r = blockIdx.x;
    if (r >= rows) return;
    float v = in[(size_t)r * HD + threadIdx.x];
    red[threadIdx.x] = v * v;
    __syncthreads();
    for (int s = 128; s > 0; s >>= 1) {
        if (threadIdx.x < s) red[threadIdx.x] += red[threadIdx.x + s];
        __syncthreads();
    }
    float n = fmaxf(sqrtf(red[0]), 1e-12f);
    out[(size_t)r * HD + threadIdx.x] = v / n;
}

__global__ void kFill(float* __restrict__ p, float v, long long n)
{
    long long i = (long long)blockIdx.x * blockDim.x + threadIdx.x;
    long long stride = (long long)gridDim.x * blockDim.x;
    for (; i < n; i += stride) p[i] = v;
}

__global__ void kCopy(float* __restrict__ d, const float* __restrict__ s, long long n)
{
    long long i = (long long)blockIdx.x * blockDim.x + threadIdx.x;
    long long stride = (long long)gridDim.x * blockDim.x;
    for (; i < n; i += stride) d[i] = s[i];
}

__global__ void kAddRelu(float* __restrict__ out, const float* __restrict__ a,
                         const float* __restrict__ b, long long n)
{
    long long i = (long long)blockIdx.x * blockDim.x + threadIdx.x;
    if (i < n) out[i] = fmaxf(a[i] + b[i], 0.0f);
}

__global__ void kGateCombine(const float* __restrict__ u, const float* __restrict__ neigh,
                             const float* __restrict__ ent, float* __restrict__ out, long long n)
{
    long long i = (long long)blockIdx.x * blockDim.x + threadIdx.x;
    if (i < n) { float g = u[i]; out[i] = g * neigh[i] + (1.0f - g) * ent[i]; }
}

__global__ void kMeanP(const float* __restrict__ F, float* __restrict__ nb, long long n)
{
    long long i = (long long)blockIdx.x * blockDim.x + threadIdx.x;
    if (i < n) nb[i] = 0.5f * (F[i] + F[(long long)NE * HD + i]);
}

// per-(row,head) 64-wide dot with attention vector
__global__ void kDotHeads(const float* __restrict__ X, const float* __restrict__ a,
                          float* __restrict__ out, int rows)
{
    int g = blockIdx.x * blockDim.x + threadIdx.x;
    if (g >= rows * NHD) return;
    int n = g >> 2, hd = g & 3;
    const float* x  = X + (size_t)n * HD + hd * DHD;
    const float* aa = a + hd * DHD;
    float s = 0.0f;
    #pragma unroll 8
    for (int d = 0; d < DHD; ++d) s += x[d] * aa[d];
    out[g] = s;
}

static __device__ __forceinline__ void atomicMaxF(float* addr, float val)
{
    unsigned* ua = (unsigned*)addr;
    unsigned old = *ua;
    while (__uint_as_float(old) < val) {
        unsigned assumed = old;
        old = atomicCAS(ua, assumed, __float_as_uint(val));
        if (old == assumed) break;
    }
}

// edge attention logits + segment max
__global__ void kEdgeAtt(const int* __restrict__ src, const int* __restrict__ dst,
                         const int* __restrict__ rid,
                         const float* __restrict__ nm, const float* __restrict__ rm,
                         const float* __restrict__ nd,
                         float* __restrict__ eatt, float* __restrict__ emax, int E)
{
    int g = blockIdx.x * blockDim.x + threadIdx.x;
    if (g >= E * NHD) return;
    int e = g >> 2, hd = g & 3;
    float v = nm[src[e] * NHD + hd] + rm[rid[e] * NHD + hd] + nd[dst[e] * NHD + hd];
    v = (v > 0.0f) ? v : 0.2f * v;     // leaky_relu 0.2
    eatt[g] = v;
    atomicMaxF(&emax[dst[e] * NHD + hd], v);
}

// segment sum of exp
__global__ void kEdgeDen(const int* __restrict__ dst, const float* __restrict__ eatt,
                         const float* __restrict__ emax, float* __restrict__ den, int E)
{
    int g = blockIdx.x * blockDim.x + threadIdx.x;
    if (g >= E * NHD) return;
    int e = g >> 2, hd = g & 3;
    atomicAdd(&den[dst[e] * NHD + hd], __expf(eatt[g] - emax[dst[e] * NHD + hd]));
}

// weighted scatter of messages: agg[dst] += alpha * (HWm[src] + RWm[rid])
__global__ void kEdgeAgg(const int* __restrict__ src, const int* __restrict__ dst,
                         const int* __restrict__ rid,
                         const float* __restrict__ eatt, const float* __restrict__ emax,
                         const float* __restrict__ den,
                         const float* __restrict__ HWm, const float* __restrict__ RWm,
                         float* __restrict__ agg, int E)
{
    long long g = (long long)blockIdx.x * blockDim.x + threadIdx.x;
    if (g >= (long long)E * HD) return;
    int e  = (int)(g >> 8);
    int x  = (int)(g & 255);
    int hd = x >> 6;
    int d  = dst[e];
    float al = __expf(eatt[e * NHD + hd] - emax[d * NHD + hd]) / (den[d * NHD + hd] + 1e-16f);
    float m  = HWm[(size_t)src[e] * HD + x] + RWm[(size_t)rid[e] * HD + x];
    atomicAdd(&agg[(size_t)d * HD + x], al * m);
}

// ---------------- ConvTransE conv stage (fused conv1d + bias + relu) ----------------
__global__ void kConvTransE(const float* __restrict__ emb, const float* __restrict__ rel,
                            const int* __restrict__ qsub, const int* __restrict__ qrel,
                            const float* __restrict__ cw, const float* __restrict__ cb,
                            float* __restrict__ out)
{
    long long g = (long long)blockIdx.x * blockDim.x + threadIdx.x;
    if (g >= (long long)QQ * CH) return;
    int q   = (int)(g / CH);
    int rem = (int)(g - (long long)q * CH);
    int c   = rem >> 8;
    int h   = rem & 255;
    const float* e = emb + (size_t)qsub[q] * HD;
    const float* r = rel + (size_t)qrel[q] * HD;
    const float* w = cw + c * 6;          // [c][2][3]
    float acc = cb[c];
    #pragma unroll
    for (int k = 0; k < 3; ++k) {
        int hh = h + k - 1;
        if (hh >= 0 && hh < HD)
            acc += e[hh] * w[k] + r[hh] * w[3 + k];
    }
    out[g] = fmaxf(acc, 0.0f);
}

// ---------------- block-local top-K (K=50) per query row ----------------
#define TKT 64
__global__ __launch_bounds__(TKT)
void kTopK(const float* __restrict__ logits, int* __restrict__ topk)
{
    __shared__ float vals[TKT * KTOP];
    __shared__ int   idxs[TKT * KTOP];
    __shared__ float rv[TKT];
    __shared__ int   ri[TKT];
    __shared__ int   winner;

    int q = blockIdx.x, t = threadIdx.x;
    const float* row = logits + (size_t)q * NE;
    float* mv = vals + t * KTOP;
    int*   mi = idxs + t * KTOP;
    for (int i = 0; i < KTOP; ++i) { mv[i] = -3.4e38f; mi[i] = 0; }

    for (int j = t; j < NE; j += TKT) {
        float v = row[j];
        if (v > mv[KTOP - 1]) {
            int p = KTOP - 1;
            while (p > 0 && mv[p - 1] < v) { mv[p] = mv[p - 1]; mi[p] = mi[p - 1]; --p; }
            mv[p] = v; mi[p] = j;
        }
    }
    __syncthreads();

    int head = 0;
    for (int r = 0; r < KTOP; ++r) {
        rv[t] = (head < KTOP) ? mv[head] : -3.4e38f;
        ri[t] = (head < KTOP) ? mi[head] : 0;
        __syncthreads();
        if (t == 0) {
            float best = rv[0]; int bt = 0;
            for (int u = 1; u < TKT; ++u) if (rv[u] > best) { best = rv[u]; bt = u; }
            topk[q * KTOP + r] = ri[bt];
            winner = bt;
        }
        __syncthreads();
        if (t == winner) head++;
        __syncthreads();
    }
}

// build candidate-graph edge lists (deterministic half/half partition)
__global__ void kBuildEdges(const int* __restrict__ qsub, const int* __restrict__ qrel,
                            const int* __restrict__ topk,
                            int* __restrict__ es, int* __restrict__ ed, int* __restrict__ er)
{
    int e = blockIdx.x * blockDim.x + threadIdx.x;
    if (e >= EC) return;
    int q = e / KTOP;
    int p = (q >= QQ / PP) ? 1 : 0;
    es[e] = qsub[q] + p * NE;
    ed[e] = topk[e] + p * NE;
    er[e] = qrel[q];
}

// ---------------- host orchestration ----------------
static inline int cdiv(long long a, long long b) { return (int)((a + b - 1) / b); }

extern "C" void kernel_launch(void* const* d_in, const int* in_sizes, int n_in,
                              void* d_out, int out_size, void* d_ws, size_t ws_size,
                              hipStream_t stream)
{
    (void)in_sizes; (void)n_in; (void)out_size; (void)ws_size;

    const int*   hist_src = (const int*)d_in[0];
    const int*   hist_rid = (const int*)d_in[1];
    const int*   hist_dst = (const int*)d_in[2];
    const int*   q_sub    = (const int*)d_in[3];
    const int*   q_rel    = (const int*)d_in[4];
    const float* ent_emb  = (const float*)d_in[5];
    const float* rel_emb  = (const float*)d_in[6];
    const float* bb_Wm    = (const float*)d_in[7];
    const float* bb_Wd    = (const float*)d_in[8];
    const float* bb_Wl    = (const float*)d_in[9];
    const float* bb_am    = (const float*)d_in[10];
    const float* bb_ad    = (const float*)d_in[11];
    const float* gate_w   = (const float*)d_in[12];
    const float* gate_b   = (const float*)d_in[13];
    const float* bb_cw    = (const float*)d_in[14];
    const float* bb_cb    = (const float*)d_in[15];
    const float* bb_fw    = (const float*)d_in[16];
    const float* bb_fb    = (const float*)d_in[17];
    const float* cp_Wm    = (const float*)d_in[18];
    const float* cp_Wd    = (const float*)d_in[19];
    const float* cp_Wl    = (const float*)d_in[20];
    const float* cp_am    = (const float*)d_in[21];
    const float* cp_ad    = (const float*)d_in[22];
    const float* cp_cw    = (const float*)d_in[23];
    const float* cp_cb    = (const float*)d_in[24];
    const float* cp_fw    = (const float*)d_in[25];
    const float* cp_fb    = (const float*)d_in[26];
    float*       out      = (float*)d_out;

    // ---- workspace carving ----
    char* w = (char*)d_ws;
    size_t off = 0;
    auto alloc = [&](size_t bytes) -> void* {
        void* p = w + off;
        off += (bytes + 255) & ~(size_t)255;
        return p;
    };
    float* ent  = (float*)alloc((size_t)NE * HD * 4);
    float* rel  = (float*)alloc((size_t)NR * HD * 4);
    float* F    = (float*)alloc((size_t)N2 * HD * 4);   // node features h
    float* T1   = (float*)alloc((size_t)N2 * HD * 4);   // HWm   (conv buffer aliases T1+T2)
    float* T2   = (float*)alloc((size_t)N2 * HD * 4);   // HWd / HWl
    float* T3   = (float*)alloc((size_t)N2 * HD * 4);   // agg
    float* NB   = (float*)alloc((size_t)NE * HD * 4);   // mean neigh
    float* RW   = (float*)alloc((size_t)NR * HD * 4);   // rel @ Wm
    float* nm   = (float*)alloc((size_t)N2 * NHD * 4);
    float* nd   = (float*)alloc((size_t)N2 * NHD * 4);
    float* rm   = (float*)alloc((size_t)NR * NHD * 4);
    float* eatt = (float*)alloc((size_t)EC * NHD * 4);
    float* emax = (float*)alloc((size_t)N2 * NHD * 4);
    float* den  = (float*)alloc((size_t)N2 * NHD * 4);
    float* feat = (float*)alloc((size_t)QQ * HD * 4);
    int*   topk = (int*)alloc((size_t)QQ * KTOP * 4);
    int*   esrc = (int*)alloc((size_t)EC * 4);
    int*   edst = (int*)alloc((size_t)EC * 4);
    int*   erid = (int*)alloc((size_t)EC * 4);
    float* CONV = T1;   // 2048*12800 floats = 104.9MB, fits in T1+T2 (163.8MB), lifetime-disjoint

    auto gemm = [&](const float* A, int lda, const float* B, int ldb, int tB,
                    float* C, int ldc, const float* bias,
                    int M, int N, int K, int act, int accm) {
        dim3 g(cdiv(N, BN), cdiv(M, BM));
        kGemmWMMA<<<g, dim3(256), 0, stream>>>(A, lda, B, ldb, tB, C, ldc, bias, M, N, K, act, accm);
    };
    auto fill = [&](float* p, float v, long long n) {
        kFill<<<dmin(65535, cdiv(n, 256)), 256, 0, stream>>>(p, v, n);
    };

    auto relgat = [&](const int* es, const int* ed, const int* er, int E, int nodes,
                      const float* Wm, const float* Wd, const float* Wl,
                      const float* am, const float* ad) {
        for (int l = 0; l < LL; ++l) {
            const float* Wml = Wm + (size_t)l * HD * HD;
            const float* Wdl = Wd + (size_t)l * HD * HD;
            const float* Wll = Wl + (size_t)l * HD * HD;
            const float* aml = am + l * HD;     // [NH,DH] = 256 floats per layer
            const float* adl = ad + l * HD;
            gemm(F, HD, Wml, HD, 0, T1, HD, nullptr, nodes, HD, HD, ACT_NONE, ACC_NONE);   // HWm
            gemm(rel, HD, Wml, HD, 0, RW, HD, nullptr, NR, HD, HD, ACT_NONE, ACC_NONE);    // RWm
            gemm(F, HD, Wdl, HD, 0, T2, HD, nullptr, nodes, HD, HD, ACT_NONE, ACC_NONE);   // HWd
            kDotHeads<<<cdiv((long long)nodes * NHD, 256), 256, 0, stream>>>(T1, aml, nm, nodes);
            kDotHeads<<<cdiv((long long)NR * NHD, 256), 256, 0, stream>>>(RW, aml, rm, NR);
            kDotHeads<<<cdiv((long long)nodes * NHD, 256), 256, 0, stream>>>(T2, adl, nd, nodes);
            fill(emax, -3.4e38f, (long long)nodes * NHD);
            fill(den, 0.0f, (long long)nodes * NHD);
            fill(T3, 0.0f, (long long)nodes * HD);
            kEdgeAtt<<<cdiv((long long)E * NHD, 256), 256, 0, stream>>>(es, ed, er, nm, rm, nd, eatt, emax, E);
            kEdgeDen<<<cdiv((long long)E * NHD, 256), 256, 0, stream>>>(ed, eatt, emax, den, E);
            kEdgeAgg<<<cdiv((long long)E * HD, 256), 256, 0, stream>>>(es, ed, er, eatt, emax, den, T1, RW, T3, E);
            gemm(F, HD, Wll, HD, 0, T2, HD, nullptr, nodes, HD, HD, ACT_NONE, ACC_NONE);   // h @ Wl
            kAddRelu<<<cdiv((long long)nodes * HD, 256), 256, 0, stream>>>(F, T3, T2, (long long)nodes * HD);
        }
    };

    // ---- phase 0: normalize embeddings ----
    kRowNorm<<<NE, 256, 0, stream>>>(ent_emb, ent, NE);
    kRowNorm<<<NR, 256, 0, stream>>>(rel_emb, rel, NR);

    // ---- phase A: backbone evolution over G history graphs ----
    for (int g = 0; g < GG; ++g) {
        kCopy<<<4096, 256, 0, stream>>>(F, ent, (long long)NE * HD);
        relgat(hist_src + g * EHE, hist_dst + g * EHE, hist_rid + g * EHE, EHE, NE,
               bb_Wm, bb_Wd, bb_Wl, bb_am, bb_ad);
        kRowNorm<<<NE, 256, 0, stream>>>(F, F, NE);                      // neigh
        // u = sigmoid([neigh, ent] @ gate_w + gate_b) as two accumulated GEMMs
        gemm(F, HD, gate_w, HD, 0, T1, HD, nullptr, NE, HD, HD, ACT_NONE, ACC_NONE);
        gemm(ent, HD, gate_w + (size_t)HD * HD, HD, 0, T1, HD, gate_b, NE, HD, HD, ACT_SIGMOID, ACC_ADD);
        kGateCombine<<<cdiv((long long)NE * HD, 256), 256, 0, stream>>>(T1, F, ent, T2, (long long)NE * HD);
        kRowNorm<<<NE, 256, 0, stream>>>(T2, ent, NE);
    }

    // ---- phase B: original logits -> d_out ----
    kConvTransE<<<cdiv((long long)QQ * CH, 256), 256, 0, stream>>>(ent, rel, q_sub, q_rel, bb_cw, bb_cb, CONV);
    gemm(CONV, CH, bb_fw, HD, 0, feat, HD, bb_fb, QQ, HD, CH, ACT_RELU, ACC_NONE);
    gemm(feat, HD, ent, HD, 1, out, NE, nullptr, QQ, NE, HD, ACT_NONE, ACC_NONE);

    // ---- phase C: top-K + candidate graph ----
    kTopK<<<QQ, TKT, 0, stream>>>(out, topk);
    kBuildEdges<<<cdiv(EC, 256), 256, 0, stream>>>(q_sub, q_rel, topk, esrc, edst, erid);

    // ---- phase D: candidate relational GAT on P*NE nodes ----
    kCopy<<<4096, 256, 0, stream>>>(F, ent, (long long)NE * HD);
    kCopy<<<4096, 256, 0, stream>>>(F + (size_t)NE * HD, ent, (long long)NE * HD);
    relgat(esrc, edst, erid, EC, N2, cp_Wm, cp_Wd, cp_Wl, cp_am, cp_ad);
    kRowNorm<<<N2, 256, 0, stream>>>(F, F, N2);
    kMeanP<<<cdiv((long long)NE * HD, 256), 256, 0, stream>>>(F, NB, (long long)NE * HD);

    // ---- phase E: enhanced logits, blended 0.5/0.5 into d_out ----
    kConvTransE<<<cdiv((long long)QQ * CH, 256), 256, 0, stream>>>(NB, rel, q_sub, q_rel, cp_cw, cp_cb, CONV);
    gemm(CONV, CH, cp_fw, HD, 0, feat, HD, cp_fb, QQ, HD, CH, ACT_RELU, ACC_NONE);
    gemm(feat, HD, NB, HD, 1, out, NE, nullptr, QQ, NE, HD, ACT_NONE, ACC_BLEND);
}
